// NeRFAll_3350074491268
// MI455X (gfx1250) — matrix operand order repaired and checked by reading, not compiled, not run on hardware
//
#include <hip/hip_runtime.h>

typedef __attribute__((ext_vector_type(16))) _Float16 v16h;
typedef __attribute__((ext_vector_type(8)))  _Float16 v8h;
typedef __attribute__((ext_vector_type(8)))  float    v8f;

#define RS 360   // LDS activation row stride in halves (63+1 | 256 | 27+5 | 8 pad)

// ---- packed weight offsets in halves (all 32B aligned) ----
static constexpr int WOFF0  = 0;        // [256][64]
static constexpr int WOFF1  = 16384;    // [256][256]
static constexpr int WOFF2  = 81920;
static constexpr int WOFF3  = 147456;
static constexpr int WOFF4  = 212992;
static constexpr int WOFF5  = 278528;   // [256][320] (skip layer)
static constexpr int WOFF6  = 360448;
static constexpr int WOFF7  = 425984;
static constexpr int WOFFFA = 491520;   // [272][256] feature(256)+alpha(1)+pad
static constexpr int WOFFV  = 561152;   // [128][288]
static constexpr int WOFFR  = 598016;   // [16][128]
static constexpr int WTOT   = 600064;
static constexpr size_t WBYTES = (size_t)WTOT * 2;

// ---- bias offsets (floats, after weights in ws) ----
static constexpr int BOFF_FA = 2048;    // 272
static constexpr int BOFF_V  = 2320;    // 128
static constexpr int BOFF_R  = 2448;    // 16

// ---- output offsets (floats) in d_out: rgb, disp, acc, weights, depth ----
static constexpr int O_RGB = 0, O_DISP = 12288, O_ACC = 16384, O_W = 20480, O_DEPTH = 282624;

// Repack fp32 weight w[k][n] (row-major, stride = fan_out) into f16 Wt[n][Kp].
// valid iff (k != skipK) && (srcK < Klim) && (n < Nlim); else 0.
__global__ void pack_w(const float* __restrict__ src, _Float16* __restrict__ dst,
                       int Nout, int Kp, int Klim, int Nlim, int stride, int skipK)
{
  int i = blockIdx.x * 256 + threadIdx.x;
  if (i >= Nout * Kp) return;
  int n = i / Kp, k = i % Kp;
  int sk = k;
  bool ok = (n < Nlim);
  if (skipK >= 0) { if (k == skipK) ok = false; else if (k > skipK) sk = k - 1; }
  float v = 0.f;
  if (ok && sk < Klim) v = src[sk * stride + n];
  dst[i] = (_Float16)v;
}

__global__ void pack_b(const float* __restrict__ src, float* __restrict__ dst, int len, int pad)
{
  int i = blockIdx.x * 256 + threadIdx.x;
  if (i < pad) dst[i] = (i < len) ? src[i] : 0.f;
}

// One MLP layer tile-set for one wave.
// MODE 0: normal store to act; MODE 1: feature+alpha (nt<16 -> act, nt==16 -> alphaB);
// MODE 2: rgb (cols 0..2 -> rgbB).
template<int TILES, int KP, bool RELU, int MODE>
__device__ __forceinline__ void mlp_layer(
    const _Float16* __restrict__ Wt, const float* __restrict__ bias,
    _Float16* act, float* alphaB, float* rgbB,
    int mt, int kcol, int ocol, int ntile0, int lane)
{
  const int lm = lane & 15;
  const int lh = lane >> 4;
  const _Float16* arow = act + (mt * 16 + lm) * RS + kcol + lh * 16;

  v8f acc[TILES];
#pragma unroll
  for (int t = 0; t < TILES; ++t) {
    float bv = bias[(ntile0 + t) * 16 + lm];
#pragma unroll
    for (int r = 0; r < 8; ++r) acc[t][r] = bv;
  }

#pragma unroll
  for (int kc = 0; kc < KP; kc += 32) {
    // A fragment (16x32 f16): lane<16 holds K kc+0..15; lane>=16 holds kc+16..31
    v8h a0 = *(const v8h*)(arow + kc);
    v8h a1 = *(const v8h*)(arow + kc + 8);
    v16h a = __builtin_shufflevector(a0, a1, 0,1,2,3,4,5,6,7,8,9,10,11,12,13,14,15);
#pragma unroll
    for (int t = 0; t < TILES; ++t) {
      // B fragment (32x16 f16): lane = N, halves = K kc..kc+15 (lanes 0-15) / kc+16..31 (lanes 16-31)
      const _Float16* bp = Wt + (size_t)((ntile0 + t) * 16 + lm) * KP + kc + lh * 16;
      v16h b = *(const v16h*)bp;
      acc[t] = __builtin_amdgcn_wmma_f32_16x16x32_f16(false, a, false, b, (short)0, acc[t], false, false);
    }
  }

  __syncthreads();   // all reads of act done before anyone overwrites it
#pragma unroll
  for (int t = 0; t < TILES; ++t) {
    int nt = ntile0 + t;
#pragma unroll
    for (int r = 0; r < 8; ++r) {
      float v = acc[t][r];
      if (RELU) v = fmaxf(v, 0.f);
      int m = mt * 16 + r + lh * 8;     // C/D layout: VGPR r, half lh -> row r + lh*8
      if (MODE == 0) {
        act[m * RS + ocol + nt * 16 + lm] = (_Float16)v;
      } else if (MODE == 1) {
        if (nt < 16)      act[m * RS + ocol + nt * 16 + lm] = (_Float16)v;
        else if (lm == 0) alphaB[m] = v;
      } else {
        if (lm < 3) rgbB[m * 3 + lm] = v;
      }
    }
  }
  __syncthreads();   // stores visible before next layer reads
}

__global__ __launch_bounds__(256) void nerf_forward(
    const float* __restrict__ rays_o, const float* __restrict__ rays_d,
    const _Float16* __restrict__ W, const float* __restrict__ Bb,
    float* __restrict__ out)
{
  __shared__ _Float16 act[64 * RS];   // 46080 B
  __shared__ float alphaB[64];
  __shared__ float rgbB[64 * 3];

  const int ray  = blockIdx.x;
  const int tid  = threadIdx.x;
  const int lane = tid & 31;
  const int wid  = tid >> 5;
  const int mt   = wid >> 1;   // M-tile (16 points each)
  const int nh   = wid & 1;    // N-half

  const float dx = rays_d[ray * 3 + 0];
  const float dy = rays_d[ray * 3 + 1];
  const float dz = rays_d[ray * 3 + 2];

  // ---- positional / view encodings ----
  if (tid < 64) {
    const float ox = rays_o[ray * 3 + 0];
    const float oy = rays_o[ray * 3 + 1];
    const float oz = rays_o[ray * 3 + 2];
    const int s = tid;
    float z  = 2.f + 4.f * ((float)s * (1.f / 63.f));
    float p0 = ox + dx * z, p1 = oy + dy * z, p2 = oz + dz * z;
    _Float16* row = act + s * RS;
    row[0] = (_Float16)p0; row[1] = (_Float16)p1; row[2] = (_Float16)p2;
#pragma unroll
    for (int j = 0; j < 10; ++j) {
      float f = (float)(1 << j);
      row[3 + 6*j + 0] = (_Float16)sinf(f * p0);
      row[3 + 6*j + 1] = (_Float16)sinf(f * p1);
      row[3 + 6*j + 2] = (_Float16)sinf(f * p2);
      row[3 + 6*j + 3] = (_Float16)cosf(f * p0);
      row[3 + 6*j + 4] = (_Float16)cosf(f * p1);
      row[3 + 6*j + 5] = (_Float16)cosf(f * p2);
    }
    row[63] = (_Float16)0.f;
    float inv = rsqrtf(dx*dx + dy*dy + dz*dz);
    float v0 = dx * inv, v1 = dy * inv, v2 = dz * inv;
    row[320] = (_Float16)v0; row[321] = (_Float16)v1; row[322] = (_Float16)v2;
#pragma unroll
    for (int j = 0; j < 4; ++j) {
      float f = (float)(1 << j);
      row[323 + 6*j + 0] = (_Float16)sinf(f * v0);
      row[323 + 6*j + 1] = (_Float16)sinf(f * v1);
      row[323 + 6*j + 2] = (_Float16)sinf(f * v2);
      row[323 + 6*j + 3] = (_Float16)cosf(f * v0);
      row[323 + 6*j + 4] = (_Float16)cosf(f * v1);
      row[323 + 6*j + 5] = (_Float16)cosf(f * v2);
    }
#pragma unroll
    for (int c = 347; c < 352; ++c) row[c] = (_Float16)0.f;
  }
  __syncthreads();

  // ---- MLP trunk (WMMA) ----
  mlp_layer<8,  64, true, 0>(W + WOFF0, Bb + 0*256, act, alphaB, rgbB, mt,  0, 64, nh*8, lane);
  mlp_layer<8, 256, true, 0>(W + WOFF1, Bb + 1*256, act, alphaB, rgbB, mt, 64, 64, nh*8, lane);
  mlp_layer<8, 256, true, 0>(W + WOFF2, Bb + 2*256, act, alphaB, rgbB, mt, 64, 64, nh*8, lane);
  mlp_layer<8, 256, true, 0>(W + WOFF3, Bb + 3*256, act, alphaB, rgbB, mt, 64, 64, nh*8, lane);
  mlp_layer<8, 256, true, 0>(W + WOFF4, Bb + 4*256, act, alphaB, rgbB, mt, 64, 64, nh*8, lane);
  mlp_layer<8, 320, true, 0>(W + WOFF5, Bb + 5*256, act, alphaB, rgbB, mt,  0, 64, nh*8, lane); // skip concat
  mlp_layer<8, 256, true, 0>(W + WOFF6, Bb + 6*256, act, alphaB, rgbB, mt, 64, 64, nh*8, lane);
  mlp_layer<8, 256, true, 0>(W + WOFF7, Bb + 7*256, act, alphaB, rgbB, mt, 64, 64, nh*8, lane);
  // feature(256) + alpha(1) fused head: 17 tiles split 9/8
  if (nh == 0) mlp_layer<9, 256, false, 1>(W + WOFFFA, Bb + BOFF_FA, act, alphaB, rgbB, mt, 64, 64, 0, lane);
  else         mlp_layer<8, 256, false, 1>(W + WOFFFA, Bb + BOFF_FA, act, alphaB, rgbB, mt, 64, 64, 9, lane);
  // views: [feat(256) | views_e(27)+pad] -> 128, relu
  mlp_layer<4, 288, true, 0>(W + WOFFV, Bb + BOFF_V, act, alphaB, rgbB, mt, 64, 64, nh*4, lane);
  // rgb: 128 -> 3 (padded to 16); both wave-halves compute identical tile
  mlp_layer<1, 128, false, 2>(W + WOFFR, Bb + BOFF_R, act, alphaB, rgbB, mt, 64, 0, 0, lane);

  // ---- volume rendering ----
  const float dn = sqrtf(dx*dx + dy*dy + dz*dz);
  if (tid < 64) {
    const int s = tid;
    float dist = ((s == 63) ? 1e10f : (4.f / 63.f)) * dn;
    float a = 1.f - expf(-fmaxf(alphaB[s], 0.f) * dist);
    alphaB[s] = a;
#pragma unroll
    for (int c = 0; c < 3; ++c)
      rgbB[s*3 + c] = 1.f / (1.f + expf(-rgbB[s*3 + c]));
  }
  __syncthreads();
  if (tid == 0) {
    float T = 1.f, accm = 0.f, depth = 0.f, c0 = 0.f, c1 = 0.f, c2 = 0.f;
    for (int s = 0; s < 64; ++s) {
      float a = alphaB[s];
      float w = a * T;
      out[O_W + ray * 64 + s] = w;
      float z = 2.f + 4.f * ((float)s * (1.f / 63.f));
      accm  += w;
      depth += w * z;
      c0 += w * rgbB[s*3 + 0];
      c1 += w * rgbB[s*3 + 1];
      c2 += w * rgbB[s*3 + 2];
      T *= (1.f - a + 1e-10f);
    }
    out[O_RGB + ray*3 + 0] = c0;
    out[O_RGB + ray*3 + 1] = c1;
    out[O_RGB + ray*3 + 2] = c2;
    out[O_ACC   + ray] = accm;
    out[O_DEPTH + ray] = depth;
    out[O_DISP  + ray] = 1.f / fmaxf(1e-10f, depth / accm);
  }
}

static inline int cdiv(int a, int b) { return (a + b - 1) / b; }

extern "C" void kernel_launch(void* const* d_in, const int* in_sizes, int n_in,
                              void* d_out, int out_size, void* d_ws, size_t ws_size,
                              hipStream_t stream) {
  const float *rays_o, *rays_d;
  const float *pw[8], *pb[8], *wf, *bf, *wa, *ba, *wv, *bvv, *wr, *br;
  auto F = [&](int i) { return (const float*)d_in[i]; };

  if (in_sizes[0] == 12288) {
    // insertion order: rays_o, rays_d, pts[0..7](w,b), feature(w,b), alpha(w,b), views(w,b), rgb(w,b)
    rays_o = F(0); rays_d = F(1);
    for (int l = 0; l < 8; ++l) { pw[l] = F(2 + 2*l); pb[l] = F(3 + 2*l); }
    wf = F(18); bf = F(19); wa = F(20); ba = F(21); wv = F(22); bvv = F(23); wr = F(24); br = F(25);
  } else {
    // jax pytree sorted keys: alpha, feature, pts, rgb, views, then rays_d, rays_o
    wa = F(0); ba = F(1); wf = F(2); bf = F(3);
    for (int l = 0; l < 8; ++l) { pw[l] = F(4 + 2*l); pb[l] = F(5 + 2*l); }
    wr = F(20); br = F(21); wv = F(22); bvv = F(23); rays_d = F(24); rays_o = F(25);
  }

  _Float16* W  = (_Float16*)d_ws;
  float*    Bf = (float*)((char*)d_ws + WBYTES);

  // ---- weight repack (f32 [k][n] -> f16 [n][Kp]) ----
  pack_w<<<cdiv(256*64, 256), 256, 0, stream>>>(pw[0], W + WOFF0, 256,  64,  63, 256, 256, -1);
  pack_w<<<cdiv(256*256,256), 256, 0, stream>>>(pw[1], W + WOFF1, 256, 256, 256, 256, 256, -1);
  pack_w<<<cdiv(256*256,256), 256, 0, stream>>>(pw[2], W + WOFF2, 256, 256, 256, 256, 256, -1);
  pack_w<<<cdiv(256*256,256), 256, 0, stream>>>(pw[3], W + WOFF3, 256, 256, 256, 256, 256, -1);
  pack_w<<<cdiv(256*256,256), 256, 0, stream>>>(pw[4], W + WOFF4, 256, 256, 256, 256, 256, -1);
  pack_w<<<cdiv(256*320,256), 256, 0, stream>>>(pw[5], W + WOFF5, 256, 320, 319, 256, 256, 63); // hole at k=63
  pack_w<<<cdiv(256*256,256), 256, 0, stream>>>(pw[6], W + WOFF6, 256, 256, 256, 256, 256, -1);
  pack_w<<<cdiv(256*256,256), 256, 0, stream>>>(pw[7], W + WOFF7, 256, 256, 256, 256, 256, -1);
  pack_w<<<cdiv(272*256,256), 256, 0, stream>>>(wf, W + WOFFFA,            272, 256, 256, 256, 256, -1);
  pack_w<<<cdiv(256,    256), 256, 0, stream>>>(wa, W + WOFFFA + 256*256,    1, 256, 256,   1,   1, -1); // alpha row
  pack_w<<<cdiv(128*288,256), 256, 0, stream>>>(wv, W + WOFFV,             128, 288, 283, 128, 128, -1);
  pack_w<<<cdiv(16*128, 256), 256, 0, stream>>>(wr, W + WOFFR,              16, 128, 128,   3,   3, -1);

  // ---- bias pack (padded, f32) ----
  for (int l = 0; l < 8; ++l)
    pack_b<<<1, 256, 0, stream>>>(pb[l], Bf + l*256, 256, 256);
  pack_b<<<2, 256, 0, stream>>>(bf, Bf + BOFF_FA,       256, 272);
  pack_b<<<1, 256, 0, stream>>>(ba, Bf + BOFF_FA + 256,   1,  16);
  pack_b<<<1, 256, 0, stream>>>(bvv, Bf + BOFF_V,       128, 128);
  pack_b<<<1, 256, 0, stream>>>(br, Bf + BOFF_R,          3,  16);

  // ---- fused NeRF forward: 1 ray per block ----
  nerf_forward<<<4096, 256, 0, stream>>>(rays_o, rays_d, W, Bf, (float*)d_out);
}
